// WPGNN_71193377899098
// MI455X (gfx1250) — compile-verified, matrix operand order checked by reading, not run on hardware
//
#include <hip/hip_runtime.h>
#include <stdint.h>

typedef _Float16 half_t;
typedef __attribute__((ext_vector_type(16))) _Float16 v16h;
typedef __attribute__((ext_vector_type(8)))  float    v8f;

#define TSTR 136  // LDS tile row stride in halves (16B-aligned, avoids 64-bank alias)

struct MLPDims { int L; int d[11]; };
// Padded concat layout: segment j of the layer-0 input lives at tile cols
// [s[j], s[j]+len[j]); ob[j] = its base row inside the original (unpadded) W.
struct SegMap { int s[4]; int len[4]; int ob[4]; int KP; };

union AFrag { v16h v; uint32_t u[8]; half_t h[16]; };
union CvtQ { uint4 q; half_t h[8]; };

__device__ __forceinline__ float leaky(float v, int act) {
  return act ? (v >= 0.f ? v : 0.01f * v) : v;
}
__device__ __forceinline__ void gadd(float* p, float v) {
  __hip_atomic_fetch_add(p, v, __ATOMIC_RELAXED, __HIP_MEMORY_SCOPE_AGENT);
}
__device__ __forceinline__ void ladd(float* p, float v) {
  __hip_atomic_fetch_add(p, v, __ATOMIC_RELAXED, __HIP_MEMORY_SCOPE_WORKGROUP);
}

// ---------------------------------------------------------------------------
// Weight packing into the wave32 WMMA B-fragment layout, with K remapped
// through SegMap (padded-K position -> original W row; gaps get zeros).
// Per-layer slot: 4 kchunks x 2 nhalves x 512 halves = 4096 halves (8 KB).
// ---------------------------------------------------------------------------
__global__ void pack_frag_kernel(const float* __restrict__ W, const float* __restrict__ b,
                                 int K, int N, SegMap sm,
                                 half_t* __restrict__ wdst, float* __restrict__ bdst) {
  const int t = threadIdx.x;
  const int KC = (K + 31) >> 5, NH = (N + 15) >> 4;
  const int total = KC * NH * 32;
  if (t < total) {
    const int lane = t & 31, fi = t >> 5;
    const int nh = fi % NH, kc = fi / NH;
    const int col = nh * 16 + (lane & 15), hf = lane >> 4;
    half_t* dst = wdst + (size_t)(kc * 2 + nh) * 512 + lane * 16;
#pragma unroll
    for (int j = 0; j < 16; ++j) {
      const int p = j >> 1, s2 = j & 1;
      const int k = kc * 32 + (p < 4 ? hf * 8 + 2 * p + s2
                                     : 16 + hf * 8 + 2 * (p - 4) + s2);
      int ko = -1;
#pragma unroll
      for (int sg = 0; sg < 4; ++sg)
        if (k >= sm.s[sg] && k < sm.s[sg] + sm.len[sg]) ko = sm.ob[sg] + (k - sm.s[sg]);
      dst[j] = (ko >= 0 && col < N) ? (half_t)W[(size_t)ko * N + col] : (half_t)0.f;
    }
  }
  if (t < 32) bdst[t] = (t < N) ? b[t] : 0.f;
}

// Plain f32 copy (stride-32 rows) for the tiny global MLP.
__global__ void pack_plain_kernel(const float* __restrict__ W, const float* __restrict__ b,
                                  int K, int N, float* __restrict__ wdst,
                                  float* __restrict__ bdst) {
  const int t = blockIdx.x * blockDim.x + threadIdx.x;
  if (t < K * 32) {
    const int k = t >> 5, n = t & 31;
    wdst[t] = (n < N) ? W[(size_t)k * N + n] : 0.f;
  }
  if (t < 32) bdst[t] = (t < N) ? b[t] : 0.f;
}

// f32 [rows, din] -> f16 [rows, 32] zero-padded (zeros matter: staging over-copies).
__global__ void cvt_pad_kernel(const float* __restrict__ in, half_t* __restrict__ out,
                               long rows, int din) {
  const long total = rows * 32;
  for (long i = (long)blockIdx.x * blockDim.x + threadIdx.x; i < total;
       i += (long)gridDim.x * blockDim.x) {
    const long r = i >> 5; const int c = (int)(i & 31);
    out[i] = (c < din) ? (half_t)in[r * din + c] : (half_t)0.f;
  }
}

__global__ void zero_f32_kernel(float* __restrict__ p, long n) {
  for (long i = (long)blockIdx.x * blockDim.x + threadIdx.x; i < n;
       i += (long)gridDim.x * blockDim.x) p[i] = 0.f;
}

// ---------------------------------------------------------------------------
// Shared MLP core: one wave computes 16 rows end-to-end with WMMA f16/f32acc.
// Layer 0 A comes from the staged LDS tile (branchless uint4 copies);
// hidden layers round-trip C->leakyrelu->f16 LDS->A in the first 32 cols.
// ---------------------------------------------------------------------------
__device__ __forceinline__ void mlp_core(
    half_t* __restrict__ tile,          // this wave's [16][TSTR] tile
    const MLPDims& md, const half_t* __restrict__ wfrag,
    const float* __restrict__ bias, int outAct,
    int lane, int row, int hf,
    v8f& c0f, v8f& c1f, int& lastNH)    // final-layer frags returned to caller
{
  const int L = md.L;
  for (int l = 0; l < L; ++l) {
    const int K = md.d[l], N = md.d[l + 1];
    const int KC = (K + 31) >> 5, NH = (N + 15) >> 4;
    v8f c0 = {}; v8f c1 = {};
    const half_t* wl = wfrag + (size_t)l * 4096;
    const uint32_t* rp = (const uint32_t*)(tile + row * TSTR);
    for (int kc = 0; kc < KC; ++kc) {
      AFrag a;
#pragma unroll
      for (int p = 0; p < 8; ++p) {
        const int kk = (p < 4) ? (hf * 8 + 2 * p) : (16 + hf * 8 + 2 * (p - 4));
        a.u[p] = rp[kc * 16 + (kk >> 1)];
      }
      {
        const v16h b0 = *(const v16h*)(wl + (size_t)(kc * 2 + 0) * 512 + lane * 16);
        c0 = __builtin_amdgcn_wmma_f32_16x16x32_f16(false, a.v, false, b0,
                                                    (short)0, c0, false, false);
      }
      if (NH > 1) {
        const v16h b1 = *(const v16h*)(wl + (size_t)(kc * 2 + 1) * 512 + lane * 16);
        c1 = __builtin_amdgcn_wmma_f32_16x16x32_f16(false, a.v, false, b1,
                                                    (short)0, c1, false, false);
      }
    }
    const bool last = (l == L - 1);
    const int doAct = (!last) || outAct;
    for (int nh = 0; nh < NH; ++nh) {
      v8f* cp = nh ? &c1 : &c0;
      const int col = nh * 16 + row;
      const float bv = (col < N) ? bias[l * 32 + col] : 0.f;
#pragma unroll
      for (int r = 0; r < 8; ++r) {
        float v = (*cp)[r] + bv;
        (*cp)[r] = leaky(v, doAct);
      }
      if (!last) {
#pragma unroll
        for (int r = 0; r < 8; ++r) {
          const int mrow = r + 8 * hf;
          tile[mrow * TSTR + col] = (half_t)(*cp)[r];
          if (NH == 1) tile[mrow * TSTR + col + 16] = (half_t)0.f;
        }
      }
    }
    if (last) { c0f = c0; c1f = c1; lastNH = NH; }
  }
}

// Zero the first KP columns of this wave's tile (uint4 stores).
__device__ __forceinline__ void tile_zero(half_t* tile, int KP, int lane) {
  const uint4 z = {0u, 0u, 0u, 0u};
  const int kp8 = KP >> 3;
#pragma unroll 4
  for (int r = 0; r < 16; ++r)
    if (lane < kp8) *(uint4*)(tile + r * TSTR + lane * 8) = z;
}

// Copy one f16 source row segment (stride-32 buffer) into the tile, 16B chunks.
__device__ __forceinline__ void seg_copy_h(half_t* tile, int row, int hf,
                                           const half_t* src, int s, int len, bool vr) {
  const int g8 = (len + 7) >> 3;
  const uint4* sp = (const uint4*)src;
  uint4* dp = (uint4*)(tile + row * TSTR + s);
  for (int c = hf; c < g8; c += 2) {
    uint4 v = {0u, 0u, 0u, 0u};
    if (vr) v = sp[c];
    dp[c] = v;
  }
}

// Copy one f32 source row segment with conversion to f16.
__device__ __forceinline__ void seg_copy_f(half_t* tile, int row, int hf,
                                           const float* src, int s, int len, bool vr) {
  const int g8 = (len + 7) >> 3;
  uint4* dp = (uint4*)(tile + row * TSTR + s);
  for (int c = hf; c < g8; c += 2) {
    CvtQ cv;
#pragma unroll
    for (int t = 0; t < 8; ++t) cv.h[t] = vr ? (half_t)src[c * 8 + t] : (half_t)0.f;
    dp[c] = cv.q;
  }
}

// ---------------------------------------------------------------------------
// Edge MLP: gather concat(x[src], x[dst], e, u[batch[src]]) -> WMMA MLP ->
// store e_out (f16) + atomic scatter into agg_dst/agg_src.
// ---------------------------------------------------------------------------
__global__ __launch_bounds__(256) void edge_mlp_kernel(
    int nE, MLPDims md, SegMap sm,
    const int* __restrict__ ei, const int* __restrict__ batch,
    const half_t* __restrict__ xin, const half_t* __restrict__ ein,
    const half_t* __restrict__ uin, half_t* __restrict__ eout,
    float* __restrict__ aggd, float* __restrict__ aggs,
    const half_t* __restrict__ wfrag, const float* __restrict__ bias,
    int outAct, int resid) {
  __shared__ half_t sTile[8][16 * TSTR];
  __shared__ int sS[8][16], sD[8][16];
  const int lane = threadIdx.x & 31, wave = threadIdx.x >> 5;
  const int row = lane & 15, hf = lane >> 4;
  const long e0 = ((long)blockIdx.x * 8 + wave) * 16;
  half_t* tile = &sTile[wave][0];

  if (hf == 0) {
    const long e = e0 + row;
    int s = 0, d = 0;
    if (e < nE) { s = ei[e]; d = ei[(long)nE + e]; }
    sS[wave][row] = s; sD[wave][row] = d;
  }
  const long myE = e0 + row;
  const bool vr = (myE < nE);
  const int myS = sS[wave][row];
  const int myD = sD[wave][row];
  const int myG = vr ? batch[myS] : 0;

  tile_zero(tile, sm.KP, lane);
  seg_copy_h(tile, row, hf, xin + (size_t)myS * 32, sm.s[0], sm.len[0], vr);
  seg_copy_h(tile, row, hf, xin + (size_t)myD * 32, sm.s[1], sm.len[1], vr);
  seg_copy_h(tile, row, hf, ein + (size_t)myE * 32, sm.s[2], sm.len[2], vr);
  seg_copy_h(tile, row, hf, uin + (size_t)myG * 32, sm.s[3], sm.len[3], vr);

  v8f c0, c1; int NH;
  mlp_core(tile, md, wfrag, bias, outAct, lane, row, hf, c0, c1, NH);

  const int N = md.d[md.L];
  for (int nh = 0; nh < NH; ++nh) {
    v8f* cp = nh ? &c1 : &c0;
    const int col = nh * 16 + row;
#pragma unroll
    for (int r = 0; r < 8; ++r) {
      const int mrow = r + 8 * hf;
      const long e = e0 + mrow;
      if (e < nE && col < N) {
        float v = (*cp)[r];
        const half_t hv = resid ? (half_t)(v + (float)ein[(size_t)e * 32 + col])
                                : (half_t)v;
        eout[(size_t)e * 32 + col] = hv;
        const float fv = (float)hv;
        gadd(&aggd[(size_t)sD[wave][mrow] * 32 + col], fv);
        gadd(&aggs[(size_t)sS[wave][mrow] * 32 + col], fv);
      }
    }
  }
}

// ---------------------------------------------------------------------------
// Node MLP: gather concat(x, agg_dst, agg_src, u[batch]) -> WMMA MLP -> x_out.
// ---------------------------------------------------------------------------
__global__ __launch_bounds__(256) void node_mlp_kernel(
    int nN, MLPDims md, SegMap sm, const int* __restrict__ batch,
    const half_t* __restrict__ xin, const float* __restrict__ aggd,
    const float* __restrict__ aggs, const half_t* __restrict__ uin,
    half_t* __restrict__ xout,
    const half_t* __restrict__ wfrag, const float* __restrict__ bias,
    int outAct, int resid) {
  __shared__ half_t sTile[8][16 * TSTR];
  const int lane = threadIdx.x & 31, wave = threadIdx.x >> 5;
  const int row = lane & 15, hf = lane >> 4;
  const long n0 = ((long)blockIdx.x * 8 + wave) * 16;
  half_t* tile = &sTile[wave][0];

  const long myN = n0 + row;
  const bool vr = (myN < nN);
  const int myG = vr ? batch[myN] : 0;

  tile_zero(tile, sm.KP, lane);
  seg_copy_h(tile, row, hf, xin  + (size_t)myN * 32, sm.s[0], sm.len[0], vr);
  seg_copy_f(tile, row, hf, aggd + (size_t)myN * 32, sm.s[1], sm.len[1], vr);
  seg_copy_f(tile, row, hf, aggs + (size_t)myN * 32, sm.s[2], sm.len[2], vr);
  seg_copy_h(tile, row, hf, uin  + (size_t)myG * 32, sm.s[3], sm.len[3], vr);

  v8f c0, c1; int NH;
  mlp_core(tile, md, wfrag, bias, outAct, lane, row, hf, c0, c1, NH);

  const int N = md.d[md.L];
  for (int nh = 0; nh < NH; ++nh) {
    v8f* cp = nh ? &c1 : &c0;
    const int col = nh * 16 + row;
#pragma unroll
    for (int r = 0; r < 8; ++r) {
      const int mrow = r + 8 * hf;
      const long n2 = n0 + mrow;
      if (n2 < nN && col < N) {
        float v = (*cp)[r];
        xout[(size_t)n2 * 32 + col] =
            resid ? (half_t)(v + (float)xin[(size_t)n2 * 32 + col]) : (half_t)v;
      }
    }
  }
}

// Per-graph segment sums, LDS-reduced per block to avoid a 64-slot atomic hotspot.
__global__ __launch_bounds__(256) void segsum_edges_kernel(
    int nE, const half_t* __restrict__ eout, const int* __restrict__ ei,
    const int* __restrict__ batch, float* __restrict__ esum, int Eout) {
  __shared__ float acc[64 * 32];
  for (int i = threadIdx.x; i < 64 * 32; i += blockDim.x) acc[i] = 0.f;
  __syncthreads();
  for (long e = (long)blockIdx.x * blockDim.x + threadIdx.x; e < nE;
       e += (long)gridDim.x * blockDim.x) {
    const int g = batch[ei[e]];
    for (int c = 0; c < Eout; ++c)
      ladd(&acc[g * 32 + c], (float)eout[(size_t)e * 32 + c]);
  }
  __syncthreads();
  for (int i = threadIdx.x; i < 64 * 32; i += blockDim.x) {
    const float v = acc[i];
    if (v != 0.f) gadd(&esum[i], v);
  }
}

__global__ __launch_bounds__(256) void segsum_nodes_kernel(
    int nN, const half_t* __restrict__ xout, const int* __restrict__ batch,
    float* __restrict__ nsum, int Nout) {
  __shared__ float acc[64 * 32];
  for (int i = threadIdx.x; i < 64 * 32; i += blockDim.x) acc[i] = 0.f;
  __syncthreads();
  for (long n = (long)blockIdx.x * blockDim.x + threadIdx.x; n < nN;
       n += (long)gridDim.x * blockDim.x) {
    const int g = batch[n];
    for (int c = 0; c < Nout; ++c)
      ladd(&acc[g * 32 + c], (float)xout[(size_t)n * 32 + c]);
  }
  __syncthreads();
  for (int i = threadIdx.x; i < 64 * 32; i += blockDim.x) {
    const float v = acc[i];
    if (v != 0.f) gadd(&nsum[i], v);
  }
}

// Tiny global MLP: one thread per graph (64 rows, negligible cost).
__global__ void glob_mlp_kernel(int nG, MLPDims md,
    const float* __restrict__ esum, const float* __restrict__ nsum,
    const half_t* __restrict__ uin, half_t* __restrict__ uout,
    const float* __restrict__ wp, const float* __restrict__ bp,
    int Eout, int Nout, int Gin, int outAct, int resid) {
  const int g = threadIdx.x;
  if (g >= nG) return;
  float fin[96], fo[32];
  const int K0 = md.d[0];
  for (int k = 0; k < K0; ++k) {
    float v;
    if (k < Eout) v = esum[g * 32 + k];
    else if (k < Eout + Nout) v = nsum[g * 32 + (k - Eout)];
    else v = (float)uin[g * 32 + (k - Eout - Nout)];
    fin[k] = v;
  }
  const int L = md.L;
  for (int l = 0; l < L; ++l) {
    const int K = md.d[l], N = md.d[l + 1];
    const int doAct = (l < L - 1) || outAct;
    const float* W = wp + (size_t)l * 4096;
    for (int n = 0; n < N; ++n) {
      float s = bp[l * 32 + n];
      for (int k = 0; k < K; ++k) s += fin[k] * W[k * 32 + n];
      fo[n] = leaky(s, doAct);
    }
    for (int n = 0; n < N; ++n) fin[n] = fo[n];
  }
  const int Gout = md.d[L];
  for (int c = 0; c < Gout; ++c) {
    float v = fin[c];
    if (resid) v += (float)uin[g * 32 + c];
    uout[g * 32 + c] = (half_t)v;
  }
}

// d_out = concat(x[100000,2], e[3200000,4], u[64,2]) as f32.
__global__ void finalize_kernel(const half_t* __restrict__ x, const half_t* __restrict__ e,
    const half_t* __restrict__ u, float* __restrict__ out,
    long nN, int dx, long nE, int de, long nG, int du) {
  const long tx = nN * dx, te = nE * de, tu = nG * du;
  const long total = tx + te + tu;
  for (long i = (long)blockIdx.x * blockDim.x + threadIdx.x; i < total;
       i += (long)gridDim.x * blockDim.x) {
    float v;
    if (i < tx) {
      const long r = i / dx; const int c = (int)(i - r * dx);
      v = (float)x[r * 32 + c];
    } else if (i < tx + te) {
      const long j = i - tx; const long r = j / de; const int c = (int)(j - r * de);
      v = (float)e[r * 32 + c];
    } else {
      const long j = i - tx - te; const long r = j / du; const int c = (int)(j - r * du);
      v = (float)u[r * 32 + c];
    }
    out[i] = v;
  }
}

static MLPDims mkdims(int K0, int hid, int out, int L) {
  MLPDims m; m.L = L; m.d[0] = K0;
  for (int i = 1; i < L; ++i) m.d[i] = hid;
  m.d[L] = out;
  for (int i = L + 1; i < 11; ++i) m.d[i] = 0;
  return m;
}
static int a8(int v) { return (v + 7) & ~7; }
static SegMap mkseg(int l0, int l1, int l2, int l3) {
  SegMap s;
  s.len[0] = l0; s.len[1] = l1; s.len[2] = l2; s.len[3] = l3;
  s.s[0] = 0; s.s[1] = a8(l0); s.s[2] = s.s[1] + a8(l1); s.s[3] = s.s[2] + a8(l2);
  s.ob[0] = 0; s.ob[1] = l0; s.ob[2] = l0 + l1; s.ob[3] = l0 + l1 + l2;
  s.KP = (s.s[3] + l3 + 31) & ~31;
  return s;
}
static SegMap idseg(int K) {
  SegMap s;
  for (int j = 0; j < 4; ++j) { s.s[j] = 0; s.len[j] = 0; s.ob[j] = 0; }
  s.len[0] = K; s.KP = (K + 31) & ~31;
  return s;
}

extern "C" void kernel_launch(void* const* d_in, const int* in_sizes, int n_in,
                              void* d_out, int out_size, void* d_ws, size_t ws_size,
                              hipStream_t stream) {
  (void)n_in; (void)out_size; (void)ws_size;
  static const int GS[6][3] = {{32,32,32},{16,16,16},{16,16,16},{8,8,8},{8,8,8},{4,2,2}};
  const float* x0    = (const float*)d_in[0];
  const int*   ei    = (const int*)d_in[1];
  const float* ea0   = (const float*)d_in[2];
  const float* u0    = (const float*)d_in[3];
  const int*   batch = (const int*)d_in[4];
  const int nN = in_sizes[0] / 3;
  const int nE = in_sizes[1] / 2;
  const int nG = in_sizes[3] / 3;

  // ---- workspace carve (total ~441 MB) ----
  size_t off = 0; char* base = (char*)d_ws;
  auto carve = [&](size_t bytes) -> void* {
    off = (off + 255) & ~(size_t)255;
    void* p = base + off; off += bytes; return p;
  };
  half_t* WFRAG = (half_t*)carve((size_t)12 * 10 * 4096 * 2);
  float*  BIASF = (float*) carve((size_t)12 * 10 * 32 * 4);
  float*  WGLOB = (float*) carve((size_t)6 * 10 * 4096 * 4);
  float*  BGLOB = (float*) carve((size_t)6 * 10 * 32 * 4);
  half_t* XA = (half_t*)carve((size_t)nN * 32 * 2);
  half_t* XB = (half_t*)carve((size_t)nN * 32 * 2);
  half_t* EA = (half_t*)carve((size_t)nE * 32 * 2);
  half_t* EB = (half_t*)carve((size_t)nE * 32 * 2);
  half_t* UA = (half_t*)carve((size_t)nG * 32 * 2);
  half_t* UB = (half_t*)carve((size_t)nG * 32 * 2);
  float* AGGD = (float*)carve((size_t)nN * 32 * 4);  // AGGD..NSUM contiguous
  float* AGGS = (float*)carve((size_t)nN * 32 * 4);
  float* ESUM = (float*)carve((size_t)nG * 32 * 4);
  float* NSUM = (float*)carve((size_t)nG * 32 * 4);

  // ---- pack all weights once (static across the launch) ----
  int pi = 6;
  for (int m = 0; m < 6; ++m) {
    const int Ein = m ? GS[m-1][0] : 2;
    const int Nin = m ? GS[m-1][1] : 3;
    const int Gin = m ? GS[m-1][2] : 3;
    const int Eout = GS[m][0], Nout = GS[m][1], Gout = GS[m][2];
    const int L = (m == 5) ? 10 : 5;
    const int hid = (m == 5) ? 10 : Eout;
    const SegMap segE = mkseg(Nin, Nin, Ein, Gin);
    const SegMap segN = mkseg(Nin, Eout, Eout, Gin);
    const int dout[3] = { Eout, Nout, Gout };
    for (int t = 0; t < 3; ++t) {
      for (int l = 0; l < L; ++l) {
        const float* W = (const float*)d_in[pi++];
        const float* B = (const float*)d_in[pi++];
        const int N = (l == L - 1) ? dout[t] : hid;
        if (t < 2) {
          const SegMap sm = (l == 0) ? (t == 0 ? segE : segN) : idseg(hid);
          const int K = (l == 0) ? sm.KP : hid;
          pack_frag_kernel<<<1, 256, 0, stream>>>(W, B, K, N, sm,
              WFRAG + ((size_t)(m * 2 + t) * 10 + l) * 4096,
              BIASF + ((size_t)(m * 2 + t) * 10 + l) * 32);
        } else {
          const int K = (l == 0) ? (Eout + Nout + Gin) : hid;
          pack_plain_kernel<<<(K * 32 + 255) / 256, 256, 0, stream>>>(W, B, K, N,
              WGLOB + ((size_t)m * 10 + l) * 4096,
              BGLOB + ((size_t)m * 10 + l) * 32);
        }
      }
    }
  }

  // ---- convert inputs to padded f16 buffers ----
  cvt_pad_kernel<<<8192, 256, 0, stream>>>(x0, XA, nN, 3);
  cvt_pad_kernel<<<8192, 256, 0, stream>>>(ea0, EA, nE, 2);
  cvt_pad_kernel<<<8, 256, 0, stream>>>(u0, UA, nG, 3);

  half_t *xa = XA, *xb = XB, *ea = EA, *eb = EB, *ua = UA, *ub = UB;
  const long zeroN = (long)nN * 64 + (long)nG * 64;  // AGGD+AGGS+ESUM+NSUM
  const int etiles = (nE + 15) / 16, eblocks = (etiles + 7) / 8;
  const int ntiles = (nN + 15) / 16, nblocks = (ntiles + 7) / 8;

  for (int m = 0; m < 6; ++m) {
    const int Ein = m ? GS[m-1][0] : 2;
    const int Nin = m ? GS[m-1][1] : 3;
    const int Gin = m ? GS[m-1][2] : 3;
    const int Eout = GS[m][0], Nout = GS[m][1], Gout = GS[m][2];
    const int L = (m == 5) ? 10 : 5;
    const int hid = (m == 5) ? 10 : Eout;
    const int outAct = (m < 5);
    const int resid = (Ein == Eout && Nin == Nout);
    const SegMap segE = mkseg(Nin, Nin, Ein, Gin);
    const SegMap segN = mkseg(Nin, Eout, Eout, Gin);
    const MLPDims de_ = mkdims(segE.KP, hid, Eout, L);
    const MLPDims dn_ = mkdims(segN.KP, hid, Nout, L);
    const MLPDims dg_ = mkdims(Eout + Nout + Gin, hid, Gout, L);

    zero_f32_kernel<<<4096, 256, 0, stream>>>(AGGD, zeroN);

    edge_mlp_kernel<<<eblocks, 256, 0, stream>>>(nE, de_, segE, ei, batch,
        xa, ea, ua, eb, AGGD, AGGS,
        WFRAG + (size_t)(m * 2 + 0) * 10 * 4096,
        BIASF + (size_t)(m * 2 + 0) * 320,
        outAct, resid);

    segsum_edges_kernel<<<512, 256, 0, stream>>>(nE, eb, ei, batch, ESUM, Eout);

    node_mlp_kernel<<<nblocks, 256, 0, stream>>>(nN, dn_, segN, batch,
        xa, AGGD, AGGS, ua, xb,
        WFRAG + (size_t)(m * 2 + 1) * 10 * 4096,
        BIASF + (size_t)(m * 2 + 1) * 320,
        outAct, resid);

    segsum_nodes_kernel<<<512, 256, 0, stream>>>(nN, xb, batch, NSUM, Nout);

    glob_mlp_kernel<<<1, 64, 0, stream>>>(nG, dg_, ESUM, NSUM, ua, ub,
        WGLOB + (size_t)m * 10 * 4096, BGLOB + (size_t)m * 320,
        Eout, Nout, Gin, outAct, resid);

    half_t* t;
    t = xa; xa = xb; xb = t;
    t = ea; ea = eb; eb = t;
    t = ua; ua = ub; ub = t;
  }

  finalize_kernel<<<8192, 256, 0, stream>>>(xa, ea, ua, (float*)d_out,
      (long)nN, GS[5][1], (long)nE, GS[5][0], (long)nG, GS[5][2]);
}
